// CausalSelfAttention_74809740362172
// MI455X (gfx1250) — compile-verified, matrix operand order checked by reading, not run on hardware
//
#include <hip/hip_runtime.h>
#include <hip/hip_bf16.h>

#define DIM     2048
#define NHEADS  16
#define HDIM    128
#define SEQ     2048
#define BATCH   2

typedef __bf16 bfloat;
typedef bfloat v4bf  __attribute__((ext_vector_type(4)));
typedef bfloat v8bf  __attribute__((ext_vector_type(8)));
typedef bfloat v16bf __attribute__((ext_vector_type(16)));
typedef float  v8f   __attribute__((ext_vector_type(8)));
typedef unsigned int u32x4 __attribute__((ext_vector_type(4)));
typedef int          i32x4 __attribute__((ext_vector_type(4)));
typedef int          i32x8 __attribute__((ext_vector_type(8)));

// ---------------------------------------------------------------------------
// float -> bf16 (round-to-nearest-even) without relying on __bf16 arithmetic.
__device__ __forceinline__ bfloat f2bf(float f) {
    union { float f; unsigned u; } a; a.f = f;
    unsigned r = a.u + 0x7FFFu + ((a.u >> 16) & 1u);
    union { unsigned short s; bfloat h; } b;
    b.s = (unsigned short)(r >> 16);
    return b.h;
}

// Load one 16x32 bf16 WMMA A/B fragment from a row-major matrix.
// Per the CDNA5 ISA layout: lane L holds row (rbase + (L&15)); lanes 0-15 hold
// K = {kc..kc+7, kc+16..kc+23}, lanes 16-31 hold K = {kc+8..kc+15, kc+24..kc+31}.
__device__ __forceinline__ v16bf load_frag_g(const bfloat* __restrict__ base,
                                             int ld, int rbase, int kc) {
    int lane = threadIdx.x & 31;
    const bfloat* p = base + (size_t)(rbase + (lane & 15)) * (size_t)ld
                           + kc + ((lane >> 4) << 3);
    v8bf lo = *(const v8bf*)p;
    v8bf hi = *(const v8bf*)(p + 16);
    return __builtin_shufflevector(lo, hi, 0,1,2,3,4,5,6,7,8,9,10,11,12,13,14,15);
}

// ---------------------------------------------------------------------------
// Tensor Data Mover: DMA a 2D tile (tile_w x tile_h elements, bf16) from a
// row-major global matrix (row stride = row_stride elements) into LDS.
// D# layout per CDNA5 ISA (group0: count/lds/global/type,
// group1: data_size=2B, dims, tile dims, dim0 stride; groups 2-3 zero => 2D).
__device__ __forceinline__ void tdm_load_2d(const bfloat* gsrc, void* lds_dst,
                                            int tile_w, int tile_h,
                                            int tensor_w, int row_stride) {
    unsigned long long ga = (unsigned long long)(size_t)gsrc;
    u32x4 g0;
    g0[0] = 1u;                                            // count=1 (user D#)
    g0[1] = (unsigned)(size_t)lds_dst;                     // LDS byte address
    g0[2] = (unsigned)ga;                                  // global_addr[31:0]
    g0[3] = (unsigned)((ga >> 32) & 0x01FFFFFFull)         // global_addr[56:32]
          | (2u << 30);                                    // type=2 ("image")

    i32x8 g1;
    g1[0] = 1 << 16;                                       // data_size=1 -> 2 bytes
    g1[1] = (tensor_w & 0xFFFF) << 16;                     // tensor_dim0[15:0]
    g1[2] = (int)((unsigned)tensor_w >> 16)                // tensor_dim0[31:16]
          | ((tile_h & 0xFFFF) << 16);                     // tensor_dim1[15:0]
    g1[3] = (tile_w & 0xFFFF) << 16;                       // tile_dim0
    g1[4] = (tile_h & 0xFFFF);                             // tile_dim1 (tile_dim2=0)
    g1[5] = row_stride;                                    // tensor_dim0_stride lo32
    g1[6] = 0;                                             // stride hi / dim1_stride
    g1[7] = 0;

    i32x4 gz = {0, 0, 0, 0};
#if defined(__clang_major__) && (__clang_major__ >= 23)
    i32x8 gz8 = {0, 0, 0, 0, 0, 0, 0, 0};
    __builtin_amdgcn_tensor_load_to_lds(g0, g1, gz, gz, gz8, 0);
#else
    __builtin_amdgcn_tensor_load_to_lds(g0, g1, gz, gz, 0);
#endif
}

// ---------------------------------------------------------------------------
// Elementwise f32 -> bf16 conversion.
__global__ void f32_to_bf16(const float* __restrict__ in, bfloat* __restrict__ out,
                            size_t n) {
    size_t i = (size_t)blockIdx.x * blockDim.x + threadIdx.x;
    if (i < n) out[i] = f2bf(in[i]);
}

// ---------------------------------------------------------------------------
// C[m,n] = sum_k A[m,k] * B[n,k]   (i.e. A @ B^T), bf16 inputs, f32 output.
// Block = 256 threads = 8 waves stacked over M; all waves share one 64-col
// B strip, so the B tile (64x32 bf16 = 4KB) is DMA'd by the TDM into a
// double-buffered LDS stage (one issue per K-step by wave 0, TENSORcnt-
// synchronized), while each wave streams its private A rows from global.
__global__ __launch_bounds__(256)
void gemm_bf16f32(const bfloat* __restrict__ A, const bfloat* __restrict__ B,
                  float* __restrict__ C, int M, int N, int K) {
    __shared__ __align__(16) bfloat bstage[2][64 * 32];    // 2 x 4KB

    int lane  = threadIdx.x & 31;
    int wave  = threadIdx.x >> 5;
    int nbase = blockIdx.x * 64;
    int mbase = blockIdx.y * 128 + wave * 16;
    int half  = lane >> 4, col = lane & 15;

    v8f zero = {0.f,0.f,0.f,0.f,0.f,0.f,0.f,0.f};
    v8f acc[4];
    #pragma unroll
    for (int f = 0; f < 4; f++) acc[f] = zero;

    const bfloat* Bblk = B + (size_t)nbase * (size_t)K;
    int T = K / 32;

    if (threadIdx.x < 32)                                  // wave 0 primes buf 0
        tdm_load_2d(Bblk, &bstage[0][0], 32, 64, K, K);

    for (int t = 0; t < T; t++) {
        int kc = t * 32;
        if (threadIdx.x < 32)
            __builtin_amdgcn_s_wait_tensorcnt(0);          // buf[t&1] landed
        __syncthreads();                                   // publish to all waves
        if (threadIdx.x < 32 && t + 1 < T)                 // overlap next DMA
            tdm_load_2d(Bblk + (kc + 32), &bstage[(t + 1) & 1][0], 32, 64, K, K);

        // prefetch the A stream ~1KB ahead (speculative, dropped if OOB)
        const bfloat* pa = A + (size_t)(mbase + col) * (size_t)K + kc;
        __builtin_prefetch(pa + 512, 0, 1);
        v16bf af = load_frag_g(A, K, mbase, kc);

        const bfloat* bp = &bstage[t & 1][0];
        #pragma unroll
        for (int f = 0; f < 4; f++) {
            const bfloat* p = bp + ((f * 16 + col) << 5) + (half << 3);
            v8bf lo = *(const v8bf*)p;
            v8bf hi = *(const v8bf*)(p + 16);
            v16bf bf = __builtin_shufflevector(lo, hi,
                           0,1,2,3,4,5,6,7,8,9,10,11,12,13,14,15);
            acc[f] = __builtin_amdgcn_wmma_f32_16x16x32_bf16(
                false, af, false, bf, (short)0, acc[f], false, false);
        }
        __syncthreads();                                   // done reading buf[t&1]
    }

    #pragma unroll
    for (int f = 0; f < 4; f++) {
        int n = nbase + f * 16 + col;
        #pragma unroll
        for (int r = 0; r < 8; r++) {
            int m = mbase + r + 8 * half;
            C[(size_t)m * (size_t)N + n] = acc[f][r];
        }
    }
}

// ---------------------------------------------------------------------------
// Per-head RMS norm + RoPE, f32 in -> bf16 out. One wave per (b,s,h) row.
__global__ __launch_bounds__(256)
void rmsnorm_rope(const float* __restrict__ in, bfloat* __restrict__ out) {
    int lane = threadIdx.x & 31;
    int wid  = blockIdx.x * (blockDim.x >> 5) + (threadIdx.x >> 5);
    int h = wid % NHEADS;
    int s = (wid / NHEADS) % SEQ;
    int b = wid / (NHEADS * SEQ);

    size_t base = ((size_t)(b * SEQ + s)) * DIM + h * HDIM;
    float4 v = *(const float4*)(in + base + lane * 4);

    float ss = v.x*v.x + v.y*v.y + v.z*v.z + v.w*v.w;
    #pragma unroll
    for (int m = 16; m >= 1; m >>= 1) ss += __shfl_xor(ss, m, 32);
    float inv = rsqrtf(ss * (1.0f / HDIM) + 1.1920928955078125e-07f);

    float nv[4] = { v.x * inv, v.y * inv, v.z * inv, v.w * inv };
    bool hi = lane >= 16;           // lanes 16-31 hold the second half (d >= 64)
    float pos = (float)s;

    v4bf o;
    #pragma unroll
    for (int j = 0; j < 4; j++) {
        int d = lane * 4 + j;
        int i = d & 63;                          // rotary pair index
        float other = __shfl_xor(nv[j], 16, 32); // partner element across halves
        float fr = pos * __powf(10000.0f, -(float)i * (1.0f / 64.0f));
        float c  = __cosf(fr), sn = __sinf(fr);
        float res = hi ? (c * nv[j] - sn * other)   // -x1*sin + x2*cos
                       : (c * nv[j] + sn * other);  //  x1*cos + x2*sin
        o[j] = f2bf(res);
    }
    *(v4bf*)(out + base + lane * 4) = o;
}

// ---------------------------------------------------------------------------
// V [b*S, DIM] f32 -> V^T [b, h, d, s] bf16 so attn*V B-fragments are contiguous.
__global__ void transpose_v(const float* __restrict__ Vf, bfloat* __restrict__ vt) {
    size_t idx = (size_t)blockIdx.x * blockDim.x + threadIdx.x;
    int d128 = (int)(idx % DIM);
    size_t t = idx / DIM;
    int s = (int)(t % SEQ);
    int b = (int)(t / SEQ);
    int h = d128 >> 7;
    int d = d128 & 127;
    vt[(((size_t)(b * NHEADS + h) * HDIM + d) * SEQ) + s] = f2bf(Vf[idx]);
}

// ---------------------------------------------------------------------------
// Flash-style causal attention. Grid = (SEQ/64, NHEADS, BATCH), 128 threads.
// Each wave owns 16 queries; streams key blocks of 32 with online softmax.
__global__ __launch_bounds__(128)
void attn_fwd(const bfloat* __restrict__ qb, const bfloat* __restrict__ kb,
              const bfloat* __restrict__ vt, bfloat* __restrict__ yb) {
    __shared__ __align__(16) bfloat plds[4][16][32];   // per-wave P staging

    int lane = threadIdx.x & 31;
    int wave = threadIdx.x >> 5;
    int half = lane >> 4, col = lane & 15;
    int h = blockIdx.y, b = blockIdx.z;
    int q0 = blockIdx.x * 64 + wave * 16;
    int rowb = b * SEQ;

    // Q fragments (16 queries x full head dim), resident for the whole kernel
    v16bf qf[4];
    #pragma unroll
    for (int c = 0; c < 4; c++)
        qf[c] = load_frag_g(qb, DIM, rowb + q0, h * HDIM + c * 32);

    v8f zero = {0.f,0.f,0.f,0.f,0.f,0.f,0.f,0.f};
    v8f o[8];
    #pragma unroll
    for (int f = 0; f < 8; f++) o[f] = zero;
    float Mr[8], Lr[8];
    #pragma unroll
    for (int r = 0; r < 8; r++) { Mr[r] = -1e30f; Lr[r] = 0.f; }

    const float scale = 0.08838834764831843f;   // 1/sqrt(128)

    for (int kv = 0; kv < q0 + 16; kv += 32) {
        // scores: two 16x16 fragments covering keys [kv..kv+31]
        v8f s0 = zero, s1 = zero;
        #pragma unroll
        for (int c = 0; c < 4; c++) {
            v16bf kf0 = load_frag_g(kb, DIM, rowb + kv,      h * HDIM + c * 32);
            v16bf kf1 = load_frag_g(kb, DIM, rowb + kv + 16, h * HDIM + c * 32);
            s0 = __builtin_amdgcn_wmma_f32_16x16x32_bf16(false, qf[c], false, kf0,
                                                         (short)0, s0, false, false);
            s1 = __builtin_amdgcn_wmma_f32_16x16x32_bf16(false, qf[c], false, kf1,
                                                         (short)0, s1, false, false);
        }

        int c0 = kv + col, c1 = c0 + 16;
        float al[8];
        #pragma unroll
        for (int r = 0; r < 8; r++) {
            int m = q0 + r + 8 * half;
            float a  = s0[r] * scale;
            float bb = s1[r] * scale;
            if (c0 > m) a  = -1e30f;            // causal mask
            if (c1 > m) bb = -1e30f;
            float mx = fmaxf(a, bb);
            #pragma unroll
            for (int msk = 8; msk >= 1; msk >>= 1)
                mx = fmaxf(mx, __shfl_xor(mx, msk, 32));   // row-max (16-lane half)
            float nm = fmaxf(Mr[r], mx);
            float alpha = __expf(Mr[r] - nm);
            Mr[r] = nm;
            float p0 = __expf(a  - nm);
            float p1 = __expf(bb - nm);
            float ps = p0 + p1;
            #pragma unroll
            for (int msk = 8; msk >= 1; msk >>= 1)
                ps += __shfl_xor(ps, msk, 32);             // row-sum
            Lr[r] = Lr[r] * alpha + ps;
            al[r] = alpha;
            // stage P (C-layout) into LDS as bf16 for A-layout reload
            plds[wave][r + 8 * half][col]      = f2bf(p0);
            plds[wave][r + 8 * half][col + 16] = f2bf(p1);
        }

        // rescale running output by per-row alpha
        #pragma unroll
        for (int f = 0; f < 8; f++)
            #pragma unroll
            for (int r = 0; r < 8; r++) o[f][r] *= al[r];

        // wave-private LDS round-trip: C-layout -> A-layout
        asm volatile("s_wait_dscnt 0x0" ::: "memory");
        const bfloat* pr = &plds[wave][col][half << 3];
        v8bf lo = *(const v8bf*)pr;
        v8bf hi = *(const v8bf*)(pr + 16);
        v16bf pa = __builtin_shufflevector(lo, hi,
                       0,1,2,3,4,5,6,7,8,9,10,11,12,13,14,15);

        // O += P @ V  (V^T rows are contiguous in s)
        #pragma unroll
        for (int f = 0; f < 8; f++) {
            v16bf vf = load_frag_g(vt, SEQ, (b * NHEADS + h) * HDIM + f * 16, kv);
            o[f] = __builtin_amdgcn_wmma_f32_16x16x32_bf16(false, pa, false, vf,
                                                           (short)0, o[f], false, false);
        }
    }

    // normalize and emit bf16 (input to the output projection GEMM)
    #pragma unroll
    for (int f = 0; f < 8; f++) {
        #pragma unroll
        for (int r = 0; r < 8; r++) {
            int m = q0 + r + 8 * half;
            float val = o[f][r] / Lr[r];
            yb[((size_t)(rowb + m)) * DIM + h * HDIM + f * 16 + col] = f2bf(val);
        }
    }
}

// ---------------------------------------------------------------------------
extern "C" void kernel_launch(void* const* d_in, const int* in_sizes, int n_in,
                              void* d_out, int out_size, void* d_ws, size_t ws_size,
                              hipStream_t stream) {
    const float* x  = (const float*)d_in[0];
    const float* Wq = (const float*)d_in[1];
    const float* Wk = (const float*)d_in[2];
    const float* Wv = (const float*)d_in[3];
    const float* Wo = (const float*)d_in[4];
    float* out = (float*)d_out;
    char*  ws  = (char*)d_ws;

    const size_t M        = (size_t)BATCH * SEQ;         // 4096 rows
    const size_t XB_BYTES = M * DIM * 2;                 // 16 MB (bf16 activations)
    const size_t WB_BYTES = (size_t)DIM * DIM * 2;       //  8 MB per bf16 weight
    const size_t F32_BYTES = M * DIM * 4;                // 32 MB f32 activations

    bfloat* xb  = (bfloat*)(ws);
    bfloat* wqb = (bfloat*)(ws + XB_BYTES);
    bfloat* wkb = wqb + (size_t)DIM * DIM;
    bfloat* wvb = wkb + (size_t)DIM * DIM;
    bfloat* wob = wvb + (size_t)DIM * DIM;
    float*  Qf  = (float*)(ws + XB_BYTES + 4 * WB_BYTES);        // 48 MB..
    float*  Kf  = Qf + M * DIM;
    float*  Vf  = Kf + M * DIM;
    bfloat* vt  = (bfloat*)((char*)Vf + F32_BYTES);              // ..ends at 160 MB
    // aliased regions (safe: producers of the aliased buffer have been consumed)
    bfloat* qbp = (bfloat*)Vf;      // Vf dead after transpose_v
    bfloat* kbp = (bfloat*)Qf;      // Qf dead after rmsnorm_rope(Q)
    bfloat* ybp = (bfloat*)Kf;      // Kf dead after rmsnorm_rope(K)

    const size_t nx = M * DIM;                 // 8,388,608
    const size_t nw = (size_t)DIM * DIM;       // 4,194,304

    // 1) bf16 conversions
    f32_to_bf16<<<(unsigned)((nx + 255) / 256), 256, 0, stream>>>(x,  xb,  nx);
    f32_to_bf16<<<(unsigned)((nw + 255) / 256), 256, 0, stream>>>(Wq, wqb, nw);
    f32_to_bf16<<<(unsigned)((nw + 255) / 256), 256, 0, stream>>>(Wk, wkb, nw);
    f32_to_bf16<<<(unsigned)((nw + 255) / 256), 256, 0, stream>>>(Wv, wvb, nw);
    f32_to_bf16<<<(unsigned)((nw + 255) / 256), 256, 0, stream>>>(Wo, wob, nw);

    // 2) QKV projections (x @ W^T)
    dim3 ggrid(DIM / 64, (unsigned)(M / 128));
    gemm_bf16f32<<<ggrid, 256, 0, stream>>>(xb, wqb, Qf, (int)M, DIM, DIM);
    gemm_bf16f32<<<ggrid, 256, 0, stream>>>(xb, wkb, Kf, (int)M, DIM, DIM);
    gemm_bf16f32<<<ggrid, 256, 0, stream>>>(xb, wvb, Vf, (int)M, DIM, DIM);

    // 3) V transpose (to [b,h,d,s] bf16)
    transpose_v<<<(unsigned)(nx / 256), 256, 0, stream>>>(Vf, vt);

    // 4) RMS norm + RoPE on Q and K -> bf16
    unsigned rope_blocks = (unsigned)(M * NHEADS / 8);   // 8 waves per block
    rmsnorm_rope<<<rope_blocks, 256, 0, stream>>>(Qf, qbp);
    rmsnorm_rope<<<rope_blocks, 256, 0, stream>>>(Kf, kbp);

    // 5) flash-style causal attention
    attn_fwd<<<dim3(SEQ / 64, NHEADS, BATCH), 128, 0, stream>>>(qbp, kbp, vt, ybp);

    // 6) output projection (y @ Wo^T) -> f32 result
    gemm_bf16f32<<<ggrid, 256, 0, stream>>>(ybp, wob, out, (int)M, DIM, DIM);
}
